// ToroidalAttentionLayer_58480274702979
// MI455X (gfx1250) — matrix operand order repaired and checked by reading, not compiled, not run on hardware
//
#include <hip/hip_runtime.h>
#include <hip/hip_bf16.h>

// ---------------------------------------------------------------------------
// ToroidalAttentionLayer for MI455X (gfx1250, wave32, WMMA + TDM).
// B=4, S=8192, D=1024, H=16, DH=64.
// Pipeline: f32->bf16 convert, 4 projection GEMMs (bf16 WMMA, f32 acc,
// TDM-fed double-buffered LDS), 3-neighbor toroidal attention, out GEMM -> f32.
// ---------------------------------------------------------------------------

typedef __attribute__((ext_vector_type(16))) __bf16        v16bf;
typedef __attribute__((ext_vector_type(4)))  __bf16        v4bf;
typedef __attribute__((ext_vector_type(8)))  float         v8f;
typedef __attribute__((ext_vector_type(4)))  unsigned int  v4u;
typedef __attribute__((ext_vector_type(8)))  int           v8i;
typedef __attribute__((ext_vector_type(4)))  int           v4i;

#define BATCH 4
#define SEQ   8192
#define DIM   1024
#define HEADS 16
#define DHEAD 64
#define ROWS  (BATCH * SEQ)   // 32768

#if defined(__gfx1250__) && __has_builtin(__builtin_amdgcn_tensor_load_to_lds)
#define USE_TDM 1
#endif

static constexpr int LDSP = 80;   // 64B tile row + 16B pad -> 20-dword stride

// ---------------------------------------------------------------------------
// f32 -> bf16 conversion (vectorized by 4)
// ---------------------------------------------------------------------------
__global__ void cvt_f32_bf16_kernel(const float* __restrict__ in,
                                    __bf16* __restrict__ out, int n4) {
    int i = blockIdx.x * blockDim.x + threadIdx.x;
    if (i < n4) {
        float4 f = ((const float4*)in)[i];
        v4bf o;
        o.x = (__bf16)f.x;
        o.y = (__bf16)f.y;
        o.z = (__bf16)f.z;
        o.w = (__bf16)f.w;
        ((v4bf*)out)[i] = o;
    }
}

// ---------------------------------------------------------------------------
// TDM: DMA a (rows x 32) bf16 tile (row stride k_ld elements) from global
// memory into LDS at byte offset lds_off with 16B padding per 64B row
// (matching LDSP=80). D# built per CDNA5 ISA sect 8.3-8.4.
// This toolchain exposes the 6-arg builtin:
//   (uint32x4 g0, int32x8 g1, int32x4 g2, int32x4 g3, int32x8, i32 cpol)
// ---------------------------------------------------------------------------
#if defined(USE_TDM)
__device__ __forceinline__ void tdm_load_tile(const __bf16* gsrc,
                                              unsigned lds_off,
                                              int rows, int k_ld) {
    unsigned long long ga = (unsigned long long)(size_t)gsrc;   // byte address
    v4u g0;
    g0.x = 1u;                                    // count=1, user descriptor
    g0.y = lds_off;                               // lds_addr (bytes)
    g0.z = (unsigned)ga;                          // global_addr[31:0]
    g0.w = (unsigned)(ga >> 32) | 0x80000000u;    // global_addr[56:32] | type=2
    v8i g1;
    // data_size=2B(1), pad_enable, pad_interval=16 dwords(3), pad_amount=4 dwords(3)
    g1[0] = (int)((1u << 16) | (1u << 20) | (3u << 22) | (3u << 25));
    g1[1] = (int)(((unsigned)k_ld & 0xFFFFu) << 16);                 // tensor_dim0 lo16
    g1[2] = (int)((((unsigned)k_ld >> 16) & 0xFFFFu) |
                  (((unsigned)rows & 0xFFFFu) << 16));               // dim0 hi / dim1 lo
    g1[3] = (int)((((unsigned)rows >> 16) & 0xFFFFu) | (32u << 16)); // dim1 hi / tile_dim0=32
    g1[4] = (int)((unsigned)rows & 0xFFFFu);                         // tile_dim1=rows, tile_dim2=0
    g1[5] = (int)(unsigned)k_ld;                                     // tensor_dim0_stride lo32
    g1[6] = 0;                                                       // stride hi / dim1_stride lo
    g1[7] = 0;
    v4i g2 = {0, 0, 0, 0};
    v4i g3 = {0, 0, 0, 0};
    v8i g4 = {0, 0, 0, 0, 0, 0, 0, 0};
    __builtin_amdgcn_tensor_load_to_lds(g0, g1, g2, g3, g4, 0);
}

template <int N>
__device__ __forceinline__ void wait_tensorcnt() {
#if __has_builtin(__builtin_amdgcn_s_wait_tensorcnt)
    __builtin_amdgcn_s_wait_tensorcnt(N);
#else
    if (N == 0) asm volatile("s_wait_tensorcnt 0x0" ::: "memory");
    else        asm volatile("s_wait_tensorcnt 0x1" ::: "memory");
#endif
}
#endif // USE_TDM

// ---------------------------------------------------------------------------
// One BK=32 K-step: 4x4 WMMA tiles per wave (64x64 wave tile).
// A (16x32, lane->M): half 0 holds K[0..7]+K[16..23], half 1 K[8..15]+K[24..31]
// B (32x16, lane->N): half 0 holds K[0..7]+K[16..23], half 1 K[8..15]+K[24..31]
// ---------------------------------------------------------------------------
__device__ __forceinline__ void compute_step(const char* sAb,
                                             const char* sBb,
                                             int waveM, int waveN,
                                             int lane15, int half,
                                             v8f (&acc)[4][4]) {
    v16bf af[4], bfr[4];
    #pragma unroll
    for (int mt = 0; mt < 4; ++mt) {
        const char* r = sAb + (waveM * 64 + mt * 16 + lane15) * LDSP + half * 16;
        ((uint4*)&af[mt])[0] = *(const uint4*)(r);
        ((uint4*)&af[mt])[1] = *(const uint4*)(r + 32);
    }
    #pragma unroll
    for (int nt = 0; nt < 4; ++nt) {
        const char* r = sBb + (waveN * 64 + nt * 16 + lane15) * LDSP + half * 16;
        ((uint4*)&bfr[nt])[0] = *(const uint4*)(r);
        ((uint4*)&bfr[nt])[1] = *(const uint4*)(r + 32);
    }
    #pragma unroll
    for (int mt = 0; mt < 4; ++mt)
        #pragma unroll
        for (int nt = 0; nt < 4; ++nt)
            acc[mt][nt] = __builtin_amdgcn_wmma_f32_16x16x32_bf16(
                false, af[mt], false, bfr[nt], (short)0, acc[mt][nt],
                false, false);
}

// ---------------------------------------------------------------------------
// GEMM: out[m,n] = (sum_k A[m,k] * W[n,k] + bias[n]) * alpha   (A @ W^T)
// Block tile 256x128, BK=32, 256 threads = 8 wave32 (4 M-waves x 2 N-waves),
// each wave owns a 64x64 tile (4x4 WMMA accumulators).
// TDM path: waves 0/1 DMA the A/B tiles into double-buffered LDS.
// ---------------------------------------------------------------------------
template <typename OutT>
__global__ __launch_bounds__(256)
void gemm_bf16_wmma_kernel(const __bf16* __restrict__ A,
                           const __bf16* __restrict__ W,
                           const float*  __restrict__ bias,
                           OutT* __restrict__ out,
                           int M, int N, int K, float alpha) {
    constexpr int BM = 256, BN = 128, BK = 32;
    constexpr unsigned offA0 = 0;
    constexpr unsigned offA1 = BM * LDSP;
    constexpr unsigned offB0 = 2 * BM * LDSP;
    constexpr unsigned offB1 = 2 * BM * LDSP + BN * LDSP;
    __shared__ __align__(16) char smem[(2 * BM + 2 * BN) * LDSP];  // 60 KB

    const int tid    = threadIdx.x;
    const int lane   = tid & 31;
    const int wave   = tid >> 5;
    const int lane15 = lane & 15;
    const int half   = lane >> 4;
    const int waveM  = wave >> 1;            // 0..3 -> 64 rows each
    const int waveN  = wave & 1;             // 0..1 -> 64 cols each
    const int blockM = blockIdx.y * BM;
    const int blockN = blockIdx.x * BN;
    const int nit    = K / BK;

    v8f acc[4][4] = {};

#if defined(USE_TDM)
    const unsigned ldsBase = (unsigned)(size_t)smem;   // LDS byte offset of smem
    const __bf16* Abase = A + (size_t)blockM * K;
    const __bf16* Wbase = W + (size_t)blockN * K;

    if (wave == 0) tdm_load_tile(Abase, ldsBase + offA0, BM, K);
    if (wave == 1) tdm_load_tile(Wbase, ldsBase + offB0, BN, K);

    for (int it = 0; it < nit; ++it) {
        const int cur = it & 1;
        if (it + 1 < nit) {
            const int k1 = (it + 1) * BK;
            if (wave == 0)
                tdm_load_tile(Abase + k1, ldsBase + (cur ? offA0 : offA1), BM, K);
            if (wave == 1)
                tdm_load_tile(Wbase + k1, ldsBase + (cur ? offB0 : offB1), BN, K);
            if (wave < 2) wait_tensorcnt<1>();   // tile `it` complete; next in flight
        } else {
            if (wave < 2) wait_tensorcnt<0>();
        }
        __syncthreads();
        const char* sAc = smem + (cur ? offA1 : offA0);
        const char* sBc = smem + (cur ? offB1 : offB0);
        compute_step(sAc, sBc, waveM, waveN, lane15, half, acc);
        __syncthreads();   // all reads done before DMA overwrites this buffer
    }
#else
    for (int it = 0; it < nit; ++it) {
        const int k0 = it * BK;
        // A tile: 256x32 bf16 = 1024 16B chunks; B tile: 128x32 = 512 chunks.
        #pragma unroll
        for (int i = 0; i < 4; ++i) {
            int c = tid + i * 256, row = c >> 2, sub = c & 3;
            *(uint4*)(smem + offA0 + row * LDSP + sub * 16) =
                *(const uint4*)(A + (size_t)(blockM + row) * K + k0 + sub * 8);
        }
        #pragma unroll
        for (int i = 0; i < 2; ++i) {
            int c = tid + i * 256, row = c >> 2, sub = c & 3;
            *(uint4*)(smem + offB0 + row * LDSP + sub * 16) =
                *(const uint4*)(W + (size_t)(blockN + row) * K + k0 + sub * 8);
        }
        __syncthreads();
        compute_step(smem + offA0, smem + offB0, waveM, waveN, lane15, half, acc);
        __syncthreads();
    }
#endif

    // ---- epilogue: bias + alpha. C/D layout: VGPR r -> M = r + 8*half ----
    #pragma unroll
    for (int nt = 0; nt < 4; ++nt) {
        int col = blockN + waveN * 64 + nt * 16 + lane15;
        float bv = bias[col];
        #pragma unroll
        for (int mt = 0; mt < 4; ++mt) {
            #pragma unroll
            for (int r = 0; r < 8; ++r) {
                int row = blockM + waveM * 64 + mt * 16 + half * 8 + r;
                float val = (acc[mt][nt][r] + bv) * alpha;
                out[(size_t)row * N + col] = (OutT)val;
            }
        }
    }
}

// ---------------------------------------------------------------------------
// Toroidal 3-neighbor attention. One block per (b,s); 16 lanes per head,
// 4 dims per lane; butterfly shuffle reduction within the 16-lane group.
// q is pre-scaled by 1/sqrt(DH) in its projection GEMM.
// ---------------------------------------------------------------------------
__global__ __launch_bounds__(256)
void toroidal_attn_kernel(const __bf16* __restrict__ q,
                          const __bf16* __restrict__ k,
                          const __bf16* __restrict__ v,
                          __bf16* __restrict__ out) {
    const int blk    = blockIdx.x;           // b*SEQ + s
    const int s      = blk & (SEQ - 1);
    const int b      = blk >> 13;            // SEQ = 2^13
    const int h      = threadIdx.x >> 4;     // 16 heads
    const int lane16 = threadIdx.x & 15;
    const int d0     = lane16 * 4;

    const size_t qoff = (((size_t)blk * HEADS) + h) * DHEAD + d0;
    v4bf qv = *(const v4bf*)(q + qoff);
    float qf0 = (float)qv.x, qf1 = (float)qv.y, qf2 = (float)qv.z, qf3 = (float)qv.w;

    const int sn[3] = { (s - 1) & (SEQ - 1), s, (s + 1) & (SEQ - 1) };
    size_t noff[3];
    float sc[3];
    #pragma unroll
    for (int j = 0; j < 3; ++j) {
        noff[j] = (((size_t)(b * SEQ + sn[j]) * HEADS) + h) * DHEAD + d0;
        v4bf kv = *(const v4bf*)(k + noff[j]);
        float p = qf0 * (float)kv.x + qf1 * (float)kv.y +
                  qf2 * (float)kv.z + qf3 * (float)kv.w;
        p += __shfl_xor(p, 1, 32);
        p += __shfl_xor(p, 2, 32);
        p += __shfl_xor(p, 4, 32);
        p += __shfl_xor(p, 8, 32);
        sc[j] = p;
    }

    float m  = fmaxf(fmaxf(sc[0], sc[1]), sc[2]);
    float e0 = __expf(sc[0] - m);
    float e1 = __expf(sc[1] - m);
    float e2 = __expf(sc[2] - m);
    float inv = 1.0f / (e0 + e1 + e2);
    float a[3] = { e0 * inv, e1 * inv, e2 * inv };

    float o0 = 0.f, o1 = 0.f, o2 = 0.f, o3 = 0.f;
    #pragma unroll
    for (int j = 0; j < 3; ++j) {
        v4bf vv = *(const v4bf*)(v + noff[j]);
        o0 += a[j] * (float)vv.x;
        o1 += a[j] * (float)vv.y;
        o2 += a[j] * (float)vv.z;
        o3 += a[j] * (float)vv.w;
    }
    v4bf ov;
    ov.x = (__bf16)o0; ov.y = (__bf16)o1; ov.z = (__bf16)o2; ov.w = (__bf16)o3;
    *(v4bf*)(out + qoff) = ov;
}

// ---------------------------------------------------------------------------
// Host-side orchestration
// ---------------------------------------------------------------------------
extern "C" void kernel_launch(void* const* d_in, const int* in_sizes, int n_in,
                              void* d_out, int out_size, void* d_ws, size_t ws_size,
                              hipStream_t stream) {
    const float* x  = (const float*)d_in[0];
    const float* Wp = (const float*)d_in[1];
    const float* bp = (const float*)d_in[2];
    const float* Wq = (const float*)d_in[3];
    const float* bq = (const float*)d_in[4];
    const float* Wk = (const float*)d_in[5];
    const float* bk = (const float*)d_in[6];
    const float* Wv = (const float*)d_in[7];
    const float* bv = (const float*)d_in[8];
    const float* Wo = (const float*)d_in[9];
    const float* bo = (const float*)d_in[10];
    float* out = (float*)d_out;
    char* ws = (char*)d_ws;

    const size_t MB = 1ull << 20;
    __bf16* Wpb = (__bf16*)(ws + 0 * MB);
    __bf16* Wqb = (__bf16*)(ws + 2 * MB);
    __bf16* Wkb = (__bf16*)(ws + 4 * MB);
    __bf16* Wvb = (__bf16*)(ws + 6 * MB);
    __bf16* Wob = (__bf16*)(ws + 8 * MB);
    __bf16* xb  = (__bf16*)(ws + 16 * MB);    // 64 MB each below
    __bf16* xpb = (__bf16*)(ws + 80 * MB);
    __bf16* qb  = (__bf16*)(ws + 144 * MB);
    __bf16* kb  = (__bf16*)(ws + 208 * MB);
    __bf16* vb  = (__bf16*)(ws + 272 * MB);
    __bf16* aob = (__bf16*)(ws + 336 * MB);

    const int nW4 = DIM * DIM / 4;            // 262144
    const int nX4 = ROWS * DIM / 4;           // 8388608

    cvt_f32_bf16_kernel<<<(nX4 + 255) / 256, 256, 0, stream>>>(x, xb, nX4);
    cvt_f32_bf16_kernel<<<(nW4 + 255) / 256, 256, 0, stream>>>(Wp, Wpb, nW4);
    cvt_f32_bf16_kernel<<<(nW4 + 255) / 256, 256, 0, stream>>>(Wq, Wqb, nW4);
    cvt_f32_bf16_kernel<<<(nW4 + 255) / 256, 256, 0, stream>>>(Wk, Wkb, nW4);
    cvt_f32_bf16_kernel<<<(nW4 + 255) / 256, 256, 0, stream>>>(Wv, Wvb, nW4);
    cvt_f32_bf16_kernel<<<(nW4 + 255) / 256, 256, 0, stream>>>(Wo, Wob, nW4);

    dim3 ggrid(DIM / 128, ROWS / 256);        // (8, 128)
    const float qscale = 0.125f;              // 1/sqrt(64)

    // xp = x @ Wp^T + bp
    gemm_bf16_wmma_kernel<__bf16><<<ggrid, 256, 0, stream>>>(
        xb, Wpb, bp, xpb, ROWS, DIM, DIM, 1.0f);
    // q = (xp @ Wq^T + bq) * scale ; k, v
    gemm_bf16_wmma_kernel<__bf16><<<ggrid, 256, 0, stream>>>(
        xpb, Wqb, bq, qb, ROWS, DIM, DIM, qscale);
    gemm_bf16_wmma_kernel<__bf16><<<ggrid, 256, 0, stream>>>(
        xpb, Wkb, bk, kb, ROWS, DIM, DIM, 1.0f);
    gemm_bf16_wmma_kernel<__bf16><<<ggrid, 256, 0, stream>>>(
        xpb, Wvb, bv, vb, ROWS, DIM, DIM, 1.0f);

    // toroidal 3-neighbor attention
    toroidal_attn_kernel<<<ROWS, 256, 0, stream>>>(qb, kb, vb, aob);

    // out = attn @ Wo^T + bo  (f32 output)
    gemm_bf16_wmma_kernel<float><<<ggrid, 256, 0, stream>>>(
        aob, Wob, bo, out, ROWS, DIM, DIM, 1.0f);
}
